// KANLayerOptimized_69157563400754
// MI455X (gfx1250) — matrix operand order repaired and checked by reading, not compiled, not run on hardware
//
#include <hip/hip_runtime.h>
#include <hip/hip_bf16.h>
#include <math.h>

// ---- CDNA5 vector types ----
typedef _Float16 v16h __attribute__((ext_vector_type(16)));
typedef _Float16 h8   __attribute__((ext_vector_type(8)));
typedef float    v8f  __attribute__((ext_vector_type(8)));
typedef unsigned int u32x4 __attribute__((ext_vector_type(4)));
typedef int          i32x4 __attribute__((ext_vector_type(4)));
typedef int          i32x8 __attribute__((ext_vector_type(8)));

#define IN_F      256
#define WIDTH     512
#define GRID_PTS  32
#define M_TILE    64
#define K_STEPS   (IN_F + IN_F / GRID_PTS)     // 256 spline + 8 skip = 264
#define TILE_ELTS (GRID_PTS * WIDTH)           // 16384 f16 per B tile (32 KB)
#define LN_EPS    1e-5f

// knots = [0,0,0,0, 1/29 .. 28/29, 1,1,1,1]  (clamped cubic, 36 knots)
__device__ __forceinline__ float knot_at(int j) {
    if (j <= 3)  return 0.0f;
    if (j >= 32) return 1.0f;
    return (float)(j - 3) * (1.0f / 29.0f);
}

// ------------------------------------------------------------------
// Prep kernel: convert spline_coeffs then skip_w to f16 in d_ws,
// laid out as 264 contiguous [32][512] K-step tiles.
// ------------------------------------------------------------------
__global__ __launch_bounds__(256)
void cvt_weights_kernel(const float* __restrict__ coeffs,
                        const float* __restrict__ skip_w,
                        _Float16* __restrict__ wout)
{
    const size_t NC = (size_t)IN_F * GRID_PTS * WIDTH;   // 4,194,304
    const size_t NS = (size_t)IN_F * WIDTH;              //   131,072
    size_t i = (size_t)blockIdx.x * blockDim.x + threadIdx.x;
    if (i < NC)          wout[i] = (_Float16)coeffs[i];
    else if (i < NC + NS) wout[i] = (_Float16)skip_w[i - NC];
}

// ------------------------------------------------------------------
// TDM: 1-D async DMA of one 32 KB f16 B-tile (global -> LDS).
// D# built per CDNA5 ISA ch.8: group0 = count/lds/global/type,
// group1 = data_size + dims.  Tracked by TENSORcnt.
// This toolchain exposes the 6-arg builtin:
//   (u32x4 g0, i32x8 g1, i32x4 g2, i32x4 g3, i32x8 g4, i32 cpol)
// ------------------------------------------------------------------
__device__ __forceinline__ void tdm_load_b_tile(const _Float16* gsrc, _Float16* ldst)
{
    unsigned long long ga = (unsigned long long)(uintptr_t)gsrc;
    unsigned int       la = (unsigned int)(uintptr_t)ldst;  // flat LDS addr: low 32 bits = LDS offset

    u32x4 g0;
    g0[0] = 1u;                                            // count=1 (valid user D#)
    g0[1] = la;                                            // lds_addr
    g0[2] = (unsigned int)(ga & 0xffffffffu);              // global_addr[31:0]
    g0[3] = (unsigned int)((ga >> 32) & 0x1ffffffu)        // global_addr[56:32]
          | (2u << 30);                                    // type=2 ("image")

    i32x8 g1;
    g1[0] = (int)(1u << 16);          // workgroup_mask=0, data_size=1 (2 bytes)
    g1[1] = (int)(16384u << 16);      // tensor_dim0[15:0]=16384 in bits[31:16]
    g1[2] = (int)(1u << 16);          // tensor_dim0 hi=0 ; tensor_dim1[15:0]=1
    g1[3] = (int)(16384u << 16);      // tensor_dim1 hi=0 ; tile_dim0=16384
    g1[4] = 0;                        // tile_dim1=0, tile_dim2=0 (1-D copy)
    g1[5] = 16384;                    // tensor_dim0_stride
    g1[6] = 0;
    g1[7] = 0;

    i32x4 g2 = {0, 0, 0, 0};
    i32x4 g3 = {0, 0, 0, 0};
    i32x8 g4 = {0, 0, 0, 0, 0, 0, 0, 0};
    __builtin_amdgcn_tensor_load_to_lds(g0, g1, g2, g3, g4, 0);
}

// ------------------------------------------------------------------
// A-tile fill: thread `row` (0..63) writes 32 f16 basis values
// (or raw input chunk for the folded-in skip GEMM).
// ------------------------------------------------------------------
__device__ __forceinline__ void fill_a_row(_Float16* arow, int s, int m0, int row,
                                           const float* __restrict__ inputs,
                                           const float* __restrict__ in_shift,
                                           const float* __restrict__ in_logscale)
{
    if (s < IN_F) {
        const int f  = s;
        float ls     = in_logscale[f];
        float sp     = (ls > 20.0f) ? ls : log1pf(expf(ls));     // softplus
        float scale  = sp + 0.001f;
        float x      = inputs[(size_t)(m0 + row) * IN_F + f];
        float xn     = 1.0f / (1.0f + expf(-(x - in_shift[f]) * scale));  // (0,1)

        int span = 3 + (int)(xn * 29.0f);                        // uniform interior 1/29
        span = (span < 3) ? 3 : ((span > 31) ? 31 : span);

        float N[4], left[4], right[4];
        N[0] = 1.0f;
        #pragma unroll
        for (int j = 1; j <= 3; ++j) {
            left[j]  = xn - knot_at(span + 1 - j);
            right[j] = knot_at(span + j) - xn;
            float saved = 0.0f;
            for (int r = 0; r < j; ++r) {
                float tmp = N[r] / (right[r + 1] + left[j - r]);
                N[r] = saved + right[r + 1] * tmp;
                saved = left[j - r] * tmp;
            }
            N[j] = saved;
        }
        const int g0i = span - 3;
        #pragma unroll
        for (int k = 0; k < GRID_PTS; ++k) arow[k] = (_Float16)0.0f;
        #pragma unroll
        for (int r = 0; r < 4; ++r) arow[g0i + r] = (_Float16)N[r];
    } else {
        const int c = s - IN_F;
        const float* src = inputs + (size_t)(m0 + row) * IN_F + c * GRID_PTS;
        #pragma unroll
        for (int k = 0; k < GRID_PTS; ++k) arow[k] = (_Float16)src[k];
    }
}

// ------------------------------------------------------------------
// Main fused kernel: GEMM (spline + skip folded into K) with TDM
// double-buffered B tiles, then bias + LayerNorm + exact GELU.
// ------------------------------------------------------------------
__global__ __launch_bounds__(512)
void kan_fused_wmma_kernel(const float* __restrict__ inputs,
                           const float* __restrict__ in_shift,
                           const float* __restrict__ in_logscale,
                           const _Float16* __restrict__ w16,    // [264][32][512] f16 tiles
                           const float* __restrict__ bias,
                           const float* __restrict__ skip_b,
                           const float* __restrict__ gamma,
                           const float* __restrict__ beta,
                           float* __restrict__ out)
{
    __shared__ __align__(128) _Float16 Blds[2][GRID_PTS][WIDTH];   // 64 KB
    __shared__ __align__(32)  _Float16 Alds[2][M_TILE][GRID_PTS];  //  8 KB
    __shared__ float Part[M_TILE][4][2];                           //  2 KB

    const int tid  = threadIdx.x;
    const int lane = tid & 31;
    const int wave = tid >> 5;
    const int wm   = wave >> 2;            // 0..3 : 16-row block
    const int wn   = wave & 3;             // 0..3 : 128-col block
    const int m0   = blockIdx.x * M_TILE;

    v8f acc[8];
    #pragma unroll
    for (int t = 0; t < 8; ++t)
        #pragma unroll
        for (int i = 0; i < 8; ++i) acc[t][i] = 0.0f;

    // ---- prologue: stage buffer 0 ----
    if (tid < M_TILE)
        fill_a_row(&Alds[0][tid][0], 0, m0, tid, inputs, in_shift, in_logscale);
    if (wave == 0) {
        tdm_load_b_tile(w16, &Blds[0][0][0]);
        __builtin_amdgcn_s_wait_tensorcnt(0);
    }
    __syncthreads();

    const int r16 = lane & 15;
    const int kb  = (lane >> 4) * 8;

    for (int s = 0; s < K_STEPS; ++s) {
        const int cur = s & 1;
        const int nxt = cur ^ 1;

        // ---- prefetch step s+1 into the other buffer ----
        if (s + 1 < K_STEPS) {
            if (tid < M_TILE)
                fill_a_row(&Alds[nxt][tid][0], s + 1, m0, tid,
                           inputs, in_shift, in_logscale);
            if (wave == 0)
                tdm_load_b_tile(w16 + (size_t)(s + 1) * TILE_ELTS, &Blds[nxt][0][0]);
        }

        // ---- compute step s: 8 x v_wmma_f32_16x16x32_f16 per wave ----
        // A frag: lanes 0-15 -> row=lane, K[0..7]+K[16..23]; lanes 16-31 -> K[8..15]+K[24..31]
        h8 alo = *reinterpret_cast<const h8*>(&Alds[cur][wm * 16 + r16][kb]);
        h8 ahi = *reinterpret_cast<const h8*>(&Alds[cur][wm * 16 + r16][kb + 16]);
        v16h afrag = __builtin_shufflevector(alo, ahi,
                       0,1,2,3,4,5,6,7,8,9,10,11,12,13,14,15);
        #pragma unroll
        for (int t = 0; t < 8; ++t) {
            // B frag: lane l holds column l%16, K[kb..kb+7]+K[kb+16..kb+23]
            v16h bfrag;
            #pragma unroll
            for (int i = 0; i < 16; ++i)
                bfrag[i] = Blds[cur][kb + (i & 7) + ((i >> 3) << 4)][wn * 128 + t * 16 + r16];
            acc[t] = __builtin_amdgcn_wmma_f32_16x16x32_f16(
                         false, afrag, false, bfrag,
                         (short)0, acc[t], false, false);
        }

        // ---- drain the async prefetch, then publish buffers ----
        if (wave == 0)
            __builtin_amdgcn_s_wait_tensorcnt(0);
        __syncthreads();
    }

    // ---------- epilogue: +bias+skip_b, LayerNorm over 512, exact GELU ----------
    float bsum[8], gam[8], bet[8];
    #pragma unroll
    for (int t = 0; t < 8; ++t) {
        int col = wn * 128 + t * 16 + (lane & 15);
        bsum[t] = bias[col] + skip_b[col];
        gam[t]  = gamma[col];
        bet[t]  = beta[col];
    }

    // C/D layout: VGPR r = (row r, lanes 0-15) and (row r+8, lanes 16-31);
    // 16-lane xor butterflies reduce both halves at once.
    #pragma unroll
    for (int r = 0; r < 8; ++r) {
        float s1 = 0.0f, s2 = 0.0f;
        #pragma unroll
        for (int t = 0; t < 8; ++t) {
            float v = acc[t][r] + bsum[t];
            s1 += v;
            s2 += v * v;
        }
        #pragma unroll
        for (int off = 8; off > 0; off >>= 1) {
            s1 += __shfl_xor(s1, off, 16);
            s2 += __shfl_xor(s2, off, 16);
        }
        if ((lane & 15) == 0) {
            int row = wm * 16 + r + ((lane >> 4) << 3);
            Part[row][wn][0] = s1;
            Part[row][wn][1] = s2;
        }
    }
    __syncthreads();

    #pragma unroll
    for (int r = 0; r < 8; ++r) {
        int row = wm * 16 + r + ((lane >> 4) << 3);
        float s1 = Part[row][0][0] + Part[row][1][0] + Part[row][2][0] + Part[row][3][0];
        float s2 = Part[row][0][1] + Part[row][1][1] + Part[row][2][1] + Part[row][3][1];
        float mu  = s1 * (1.0f / 512.0f);
        float var = s2 * (1.0f / 512.0f) - mu * mu;
        float inv = rsqrtf(var + LN_EPS);
        #pragma unroll
        for (int t = 0; t < 8; ++t) {
            float v = acc[t][r] + bsum[t];
            float n = (v - mu) * inv * gam[t] + bet[t];
            float g = 0.5f * n * (1.0f + erff(n * 0.70710678118654752f));
            int col = wn * 128 + t * 16 + (lane & 15);
            out[(size_t)(m0 + row) * WIDTH + col] = g;
        }
    }
}

extern "C" void kernel_launch(void* const* d_in, const int* in_sizes, int n_in,
                              void* d_out, int out_size, void* d_ws, size_t ws_size,
                              hipStream_t stream) {
    const float* inputs   = (const float*)d_in[0];
    const float* shift    = (const float*)d_in[1];
    const float* logscale = (const float*)d_in[2];
    const float* coeffs   = (const float*)d_in[3];
    const float* bias     = (const float*)d_in[4];
    const float* skip_w   = (const float*)d_in[5];
    const float* skip_b   = (const float*)d_in[6];
    const float* gamma    = (const float*)d_in[7];
    const float* beta     = (const float*)d_in[8];
    float*     out = (float*)d_out;
    _Float16*  w16 = (_Float16*)d_ws;    // needs 264*16384*2 = 8.25 MB scratch

    // 1) convert weights to f16 K-step tiles in d_ws
    {
        const size_t total = (size_t)K_STEPS * TILE_ELTS;   // 4,325,376
        dim3 grid((unsigned)((total + 255) / 256));
        dim3 block(256);
        hipLaunchKernelGGL(cvt_weights_kernel, grid, block, 0, stream,
                           coeffs, skip_w, w16);
    }
    // 2) fused TDM + WMMA + LayerNorm + GELU kernel
    {
        dim3 grid(8192 / M_TILE);   // 128 blocks of 64 batch rows
        dim3 block(512);            // 16 wave32s
        hipLaunchKernelGGL(kan_fused_wmma_kernel, grid, block, 0, stream,
                           inputs, shift, logscale, w16, bias,
                           skip_b, gamma, beta, out);
    }
}
